// DendriticResidualModel_88029649699012
// MI455X (gfx1250) — matrix operand order repaired and checked by reading, not compiled, not run on hardware
//
#include <hip/hip_runtime.h>

typedef _Float16 v16h __attribute__((ext_vector_type(16)));
typedef _Float16 v8h  __attribute__((ext_vector_type(8)));
typedef float    v8f  __attribute__((ext_vector_type(8)));

#define T_STEPS   128
#define K_TOTAL   384          // 256 (x/Wa) + 64 (input/Wi) + 64 (t_emb/Wt)
#define K_STEP    32
#define NSTEPS    (K_TOTAL / K_STEP)   // 12
#define M_TILE    128
#define LDA       40           // 32 halves + pad; multiple of 8 -> 16B-aligned rows
#define LDB       40
#define LDS_ACT   68           // 64 floats + pad to break bank conflicts

#define A_HALVES  ((size_t)2048 * K_TOTAL)    // packed f16 A_cat
#define W_HALVES  ((size_t)16384 * K_TOTAL)   // packed f16 W_cat
#define TP_FLOATS ((size_t)128 * (4096 + 1024 + 256))

__device__ __forceinline__ float softplusf(float x) {
  // log1p(exp(x)), numerically stable
  return fmaxf(x, 0.f) + log1pf(__expf(-fabsf(x)));
}

// ---------------------------------------------------------------------------
// tproj3[t,q]=t_emb[t]·tW3[q]+tb3[q] (q<4096), tproj2 (q<1024), tproj1 (q<256)
// ---------------------------------------------------------------------------
__global__ __launch_bounds__(256) void tproj_kernel(
    const float* __restrict__ te,
    const float* __restrict__ tW3, const float* __restrict__ tb3,
    const float* __restrict__ tW2, const float* __restrict__ tb2,
    const float* __restrict__ tW1, const float* __restrict__ tb1,
    float* __restrict__ tp3, float* __restrict__ tp2, float* __restrict__ tp1) {
  const int per_t = 4096 + 1024 + 256;  // 5376
  int idx = blockIdx.x * blockDim.x + threadIdx.x;
  if (idx >= T_STEPS * per_t) return;
  int t = idx / per_t;
  int r = idx - t * per_t;
  const float* w; const float* b; float* o; int q;
  if (r < 4096)      { q = r;        w = tW3 + (size_t)q * 64; b = tb3; o = tp3 + (size_t)t * 4096 + q; }
  else if (r < 5120) { q = r - 4096; w = tW2 + (size_t)q * 64; b = tb2; o = tp2 + (size_t)t * 1024 + q; }
  else               { q = r - 5120; w = tW1 + (size_t)q * 64; b = tb1; o = tp1 + (size_t)t * 256  + q; }
  const float* e = te + (size_t)t * 64;
  float s = b[q];
  #pragma unroll
  for (int d = 0; d < 64; ++d) s = fmaf(e[d], w[d], s);
  *o = s;
}

// ---------------------------------------------------------------------------
// One-time f32 -> f16 packing of A_cat = [x | input | t_emb] (2048 x 384) and
// W_cat = [Wa | Wi | Wt] (16384 x 384). 8 halves per thread.
// ---------------------------------------------------------------------------
__global__ __launch_bounds__(256) void pack_f16_kernel(
    const float* __restrict__ x,  const float* __restrict__ te,
    const float* __restrict__ iv,
    const float* __restrict__ Wa, const float* __restrict__ Wi,
    const float* __restrict__ Wt,
    _Float16* __restrict__ Ah, _Float16* __restrict__ Wh) {
  const size_t idx = ((size_t)blockIdx.x * blockDim.x + threadIdx.x) * 8;
  if (idx >= A_HALVES + W_HALVES) return;
  const float* src; _Float16* dst;
  if (idx < A_HALVES) {
    const int m = (int)(idx / K_TOTAL), k = (int)(idx % K_TOTAL);
    if (k < 256)      src = x  + (size_t)m * 256 + k;
    else if (k < 320) src = iv + (size_t)m * 64  + (k - 256);
    else              src = te + (size_t)(m & (T_STEPS - 1)) * 64 + (k - 320);
    dst = Ah + idx;
  } else {
    const size_t w = idx - A_HALVES;
    const int c = (int)(w / K_TOTAL), k = (int)(w % K_TOTAL);
    if (k < 256)      src = Wa + (size_t)c * 256 + k;
    else if (k < 320) src = Wi + (size_t)c * 64  + (k - 256);
    else              src = Wt + (size_t)c * 64  + (k - 320);
    dst = Wh + w;
  }
  const float4 v0 = reinterpret_cast<const float4*>(src)[0];
  const float4 v1 = reinterpret_cast<const float4*>(src)[1];
  dst[0] = (_Float16)v0.x; dst[1] = (_Float16)v0.y;
  dst[2] = (_Float16)v0.z; dst[3] = (_Float16)v0.w;
  dst[4] = (_Float16)v1.x; dst[5] = (_Float16)v1.y;
  dst[6] = (_Float16)v1.z; dst[7] = (_Float16)v1.w;
}

// ---------------------------------------------------------------------------
// Fused fanout-GEMM (WMMA f16 -> f32) + softplus + 3-level dendritic tree.
// grid = (256 node-groups, 16 row-tiles), block = 256 threads = 8 waves.
// Block tile: 128 rows x 64 cols (one dendrite group n).
// Wave (wrg = w>>2, wc = w&3) owns 4 row-subtiles x 1 col-subtile,
// double-buffered LDS ping-pong with register prefetch of the next K-tile.
// F16SRC=true: operands come pre-packed f16 from workspace (no cvt in loop).
// ---------------------------------------------------------------------------
template <bool F16SRC>
__global__ __launch_bounds__(256) void dendritic_main_kernel(
    const float* __restrict__ x,  const float* __restrict__ te,
    const float* __restrict__ iv,
    const float* __restrict__ Wa, const float* __restrict__ ba,
    const float* __restrict__ Wi, const float* __restrict__ Wt,
    const _Float16* __restrict__ Ah, const _Float16* __restrict__ Wh,
    const float* __restrict__ w3, const float* __restrict__ w2,
    const float* __restrict__ w1,
    const float* __restrict__ tp3, const float* __restrict__ tp2,
    const float* __restrict__ tp1,
    float* __restrict__ out) {
  __shared__ __align__(16) _Float16 Alds[2][M_TILE * LDA];
  __shared__ __align__(16) _Float16 Blds[2][64 * LDB];
  __shared__ __align__(16) float    acts[M_TILE * LDS_ACT];
  __shared__ float red[M_TILE * 4];

  const int th      = threadIdx.x;
  const int n       = blockIdx.x;            // dendrite group 0..255
  const int colbase = n * 64;
  const int rowbase = blockIdx.y * M_TILE;

  const int wave = th >> 5;
  const int lane = th & 31;
  const int wrg = wave >> 2;                 // row half 0..1 (4 subtiles each)
  const int wc  = wave & 3;                  // col subtile 0..3
  const int lm  = lane & 15;
  const int lh  = lane >> 4;

  // cooperative tile-load indices
  const int a_row = th >> 1, a_k = (th & 1) * 16;  // A tile: 128 rows x 32 k, 16 halves/thread
  const int b_col = th >> 2, b_k = (th & 3) * 8;   // B tile:  64 cols x 32 k,  8 halves/thread
  const int a_m = rowbase + a_row;
  const int b_c = colbase + b_col;

  float a_pf[16];
  float b_pf[8];
  v8h  a_pfh[2];
  v8h  b_pfh;

  // ---- prefetch (global -> registers) ----
  auto load_tile = [&](int kbase) {
    if constexpr (F16SRC) {
      const _Float16* asrc = Ah + (size_t)a_m * K_TOTAL + kbase + a_k;
      a_pfh[0] = reinterpret_cast<const v8h*>(asrc)[0];
      a_pfh[1] = reinterpret_cast<const v8h*>(asrc)[1];
      b_pfh = *reinterpret_cast<const v8h*>(Wh + (size_t)b_c * K_TOTAL + kbase + b_k);
    } else {
      const float* asrc;
      if (kbase < 256)      asrc = x  + (size_t)a_m * 256 + (kbase + a_k);
      else if (kbase < 320) asrc = iv + (size_t)a_m * 64  + (kbase - 256 + a_k);
      else                  asrc = te + (size_t)(a_m & (T_STEPS - 1)) * 64 + (kbase - 320 + a_k);
      #pragma unroll
      for (int u = 0; u < 4; ++u) {
        const float4 v = reinterpret_cast<const float4*>(asrc)[u];
        a_pf[4 * u + 0] = v.x; a_pf[4 * u + 1] = v.y;
        a_pf[4 * u + 2] = v.z; a_pf[4 * u + 3] = v.w;
      }
      const float* bsrc;
      if (kbase < 256)      bsrc = Wa + (size_t)b_c * 256 + (kbase + b_k);
      else if (kbase < 320) bsrc = Wi + (size_t)b_c * 64  + (kbase - 256 + b_k);
      else                  bsrc = Wt + (size_t)b_c * 64  + (kbase - 320 + b_k);
      #pragma unroll
      for (int u = 0; u < 2; ++u) {
        const float4 v = reinterpret_cast<const float4*>(bsrc)[u];
        b_pf[4 * u + 0] = v.x; b_pf[4 * u + 1] = v.y;
        b_pf[4 * u + 2] = v.z; b_pf[4 * u + 3] = v.w;
      }
    }
  };
  // ---- registers -> LDS ----
  auto store_tile = [&](int q) {
    if constexpr (F16SRC) {
      v8h* ad = reinterpret_cast<v8h*>(&Alds[q][a_row * LDA + a_k]);
      ad[0] = a_pfh[0];
      ad[1] = a_pfh[1];
      *reinterpret_cast<v8h*>(&Blds[q][b_col * LDB + b_k]) = b_pfh;
    } else {
      _Float16* ad = &Alds[q][a_row * LDA + a_k];
      #pragma unroll
      for (int e = 0; e < 16; ++e) ad[e] = (_Float16)a_pf[e];
      _Float16* bd = &Blds[q][b_col * LDB + b_k];
      #pragma unroll
      for (int e = 0; e < 8; ++e) bd[e] = (_Float16)b_pf[e];
    }
  };

  v8f acc[4] = {};

  load_tile(0);
  store_tile(0);
  __syncthreads();

  #pragma unroll
  for (int kk = 0; kk < NSTEPS; ++kk) {
    const int p = kk & 1;
    if (kk + 1 < NSTEPS) load_tile((kk + 1) * K_STEP);

    // B fragment (32x16): lane holds col N=lm; K = 16*lh .. 16*lh+15
    v16h bf;
    {
      const int br = (16 * wc + lm) * LDB + 16 * lh;
      const v8h b0 = *reinterpret_cast<const v8h*>(&Blds[p][br]);
      const v8h b1 = *reinterpret_cast<const v8h*>(&Blds[p][br + 8]);
      #pragma unroll
      for (int e = 0; e < 8; ++e) { bf[e] = b0[e]; bf[8 + e] = b1[e]; }
    }
    // 4 row-subtiles share one B fragment
    #pragma unroll
    for (int p4 = 0; p4 < 4; ++p4) {
      // A fragment (16x32): lane holds row M=lm; K = 16*lh .. 16*lh+15
      const int ar = (64 * wrg + 16 * p4 + lm) * LDA;
      const v8h alo = *reinterpret_cast<const v8h*>(&Alds[p][ar + 16 * lh]);
      const v8h ahi = *reinterpret_cast<const v8h*>(&Alds[p][ar + 16 * lh + 8]);
      v16h af;
      #pragma unroll
      for (int e = 0; e < 8; ++e) { af[e] = alo[e]; af[8 + e] = ahi[e]; }
      acc[p4] = __builtin_amdgcn_wmma_f32_16x16x32_f16(
          false, af, false, bf, (short)0, acc[p4], false, false);
    }
    if (kk + 1 < NSTEPS) store_tile(1 - p);
    __syncthreads();
  }

  // ---- bias + softplus; stash activated fanout tile ----
  {
    const int c_local = 16 * wc + lm;
    const float bias = ba[colbase + c_local];
    #pragma unroll
    for (int p4 = 0; p4 < 4; ++p4) {
      #pragma unroll
      for (int v = 0; v < 8; ++v) {
        const int r_local = 64 * wrg + 16 * p4 + v + 8 * lh;  // C: M=v (+8 hi lanes)
        acts[r_local * LDS_ACT + c_local] = softplusf(acc[p4][v] + bias);
      }
    }
  }
  __syncthreads();

  // ---- tree levels 3 -> 2 : 512 (row,i) items over 256 threads ----
  #pragma unroll
  for (int rep = 0; rep < 2; ++rep) {
    const int item = th + 256 * rep;
    const int row = item >> 2, i = item & 3;
    const int t = (rowbase + row) & (T_STEPS - 1);
    float a3[4];
    #pragma unroll
    for (int j = 0; j < 4; ++j) {
      float s = 0.f;
      #pragma unroll
      for (int k = 0; k < 4; ++k)
        s = fmaf(w3[n * 64 + i * 16 + j * 4 + k],
                 acts[row * LDS_ACT + i * 16 + j * 4 + k], s);
      a3[j] = softplusf(s + tp3[(size_t)t * 4096 + n * 16 + i * 4 + j]);
    }
    float s2 = 0.f;
    #pragma unroll
    for (int j = 0; j < 4; ++j) s2 = fmaf(w2[n * 16 + i * 4 + j], a3[j], s2);
    red[row * 4 + i] = softplusf(s2 + tp2[(size_t)t * 1024 + n * 4 + i]);
  }
  __syncthreads();

  // ---- tree level 1 -> output [B*T, 256] ----
  if (th < M_TILE) {
    const int row = th;
    const int m = rowbase + row;
    const int t = m & (T_STEPS - 1);
    float s1 = 0.f;
    #pragma unroll
    for (int i = 0; i < 4; ++i) s1 = fmaf(w1[n * 4 + i], red[row * 4 + i], s1);
    out[(size_t)m * 256 + n] = softplusf(s1 + tp1[(size_t)t * 256 + n]);
  }
}

// ---------------------------------------------------------------------------
extern "C" void kernel_launch(void* const* d_in, const int* in_sizes, int n_in,
                              void* d_out, int out_size, void* d_ws, size_t ws_size,
                              hipStream_t stream) {
  const float* x   = (const float*)d_in[0];
  const float* te  = (const float*)d_in[1];
  const float* iv  = (const float*)d_in[2];
  const float* Wa  = (const float*)d_in[3];
  const float* ba  = (const float*)d_in[4];
  const float* Wt  = (const float*)d_in[5];
  const float* Wi  = (const float*)d_in[6];
  const float* w3  = (const float*)d_in[7];
  const float* tW3 = (const float*)d_in[8];
  const float* tb3 = (const float*)d_in[9];
  const float* w2  = (const float*)d_in[10];
  const float* tW2 = (const float*)d_in[11];
  const float* tb2 = (const float*)d_in[12];
  const float* w1  = (const float*)d_in[13];
  const float* tW1 = (const float*)d_in[14];
  const float* tb1 = (const float*)d_in[15];
  float* out = (float*)d_out;

  float* tp3 = (float*)d_ws;                 // [128, 4096]
  float* tp2 = tp3 + (size_t)128 * 4096;     // [128, 1024]
  float* tp1 = tp2 + (size_t)128 * 1024;     // [128, 256]

  const int tproj_total = T_STEPS * (4096 + 1024 + 256);
  tproj_kernel<<<(tproj_total + 255) / 256, 256, 0, stream>>>(
      te, tW3, tb3, tW2, tb2, tW1, tb1, tp3, tp2, tp1);

  const size_t ws_needed_f16 = TP_FLOATS * 4 + (A_HALVES + W_HALVES) * 2;
  dim3 grid(256, 2048 / M_TILE);  // 256 node groups x 16 row tiles

  if (ws_size >= ws_needed_f16) {
    _Float16* Ah = (_Float16*)((char*)d_ws + TP_FLOATS * 4);
    _Float16* Wh = Ah + A_HALVES;
    const size_t chunks = (A_HALVES + W_HALVES) / 8;
    pack_f16_kernel<<<(int)((chunks + 255) / 256), 256, 0, stream>>>(
        x, te, iv, Wa, Wi, Wt, Ah, Wh);
    dendritic_main_kernel<true><<<grid, 256, 0, stream>>>(
        x, te, iv, Wa, ba, Wi, Wt, Ah, Wh, w3, w2, w1, tp3, tp2, tp1, out);
  } else {
    dendritic_main_kernel<false><<<grid, 256, 0, stream>>>(
        x, te, iv, Wa, ba, Wi, Wt, nullptr, nullptr, w3, w2, w1, tp3, tp2, tp1, out);
  }
}
